// VQEmbedding_52999896432727
// MI455X (gfx1250) — matrix operand order repaired and checked by reading, not compiled, not run on hardware
//
#include <hip/hip_runtime.h>

// VQ nearest-codebook (CDNA5 / gfx1250, wave32, WMMA bf16 path)
//
// X[32768,512] fp32 vs codebook CB[8192,512] fp32.
// score[m,k] = ||c_k||^2 + 2*x.c  (reference's +2 alpha; ||x||^2 row-constant,
// dropped for argmin). Outputs: z_q_x f32, zqx_tilde f32, idx (as float).
//
// 275 GFLOP GEMM, compute bound -> V_WMMA_F32_16X16X32_BF16.
// Each wave holds 32 M-rows of A resident in VGPRs so every LDS B-fragment
// feeds TWO wmmas (halves LDS bytes/FLOP). Codebook chunks are staged into
// double-buffered LDS with global_load_async_to_lds_b128 (ASYNCcnt).
// B-fragments use an explicit 4-slot rotating register buffer so 3 DS
// double-loads stay in flight ahead of every wmma pair (no dscnt-0 drains).

#define BN_TOTAL 32768
#define CDIM     512
#define KDIM     8192
#define MTILE    256                  // rows per workgroup (8 waves x 32)
#define NCHUNK   128                  // codebook rows per LDS buffer
#define NCHUNKS  (KDIM / NCHUNK)      // 64
#define LDSB_ELEMS (NCHUNK * CDIM)    // ushorts per B buffer

#define LDS_AS __attribute__((address_space(3)))

typedef __attribute__((ext_vector_type(16))) __bf16 v16bf;
typedef __attribute__((ext_vector_type(8)))  float  v8f;
typedef __attribute__((ext_vector_type(4)))  int    v4i;

union FragBF {
    v16bf v;
    unsigned short s[16];
};

#if defined(__has_builtin)
#if __has_builtin(__builtin_amdgcn_global_load_async_to_lds_b128) && \
    __has_builtin(__builtin_amdgcn_s_wait_asynccnt)
#define USE_ASYNC_COPY 1
#endif
#endif

__device__ __forceinline__ unsigned short f2bf_rne(float f) {
    unsigned u = __builtin_bit_cast(unsigned, f);
    u += 0x7FFFu + ((u >> 16) & 1u);   // round-to-nearest-even
    return (unsigned short)(u >> 16);
}

// Issue one 128KB chunk copy (32 x b128 per thread). Async when available.
__device__ __forceinline__ void issue_chunk_copy(const v4i* __restrict__ src,
                                                 v4i* __restrict__ dstLds,
                                                 int tid)
{
#ifdef USE_ASYNC_COPY
    typedef LDS_AS v4i lv4i;                              // LDS
    typedef __attribute__((address_space(1))) v4i gv4i;   // global
#pragma unroll
    for (int i = 0; i < 32; ++i) {
        __builtin_amdgcn_global_load_async_to_lds_b128(
            (gv4i*)(void*)(src + tid + i * 256),
            (lv4i*)(void*)(dstLds + tid + i * 256),
            0, 0);
    }
#else
    // Plain ext-vector copy: assignment is legal through an AS(3) lvalue.
    LDS_AS v4i* d = (LDS_AS v4i*)(void*)dstLds;
#pragma unroll
    for (int i = 0; i < 32; ++i)
        d[tid + i * 256] = src[tid + i * 256];
#endif
}

__device__ __forceinline__ void wait_chunk_copy(bool morePending)
{
#ifdef USE_ASYNC_COPY
    // Async loads complete in order: waiting <=32 guarantees the previous
    // chunk's 32 loads landed, while chunk i+1's 32 stay in flight.
    if (morePending) __builtin_amdgcn_s_wait_asynccnt(32);
    else             __builtin_amdgcn_s_wait_asynccnt(0);
#endif
}

// ---------------------------------------------------------------------------
// Phase 1: codebook fp32 -> bf16 + per-row squared norms. One wave per row.
// ---------------------------------------------------------------------------
__global__ __launch_bounds__(256)
void vq_prep_cb(const float* __restrict__ CB,
                unsigned short* __restrict__ CBbf,
                float* __restrict__ cbsq)
{
    const int row  = (blockIdx.x * blockDim.x + threadIdx.x) >> 5;
    const int lane = threadIdx.x & 31;
    if (row >= KDIM) return;

    const float* src = CB + (size_t)row * CDIM + lane * 16;
    unsigned short* dst = CBbf + (size_t)row * CDIM + lane * 16;

    union { unsigned short s[16]; uint4 q[2]; } pk;
    float sum = 0.0f;
#pragma unroll
    for (int j = 0; j < 4; ++j) {
        float4 f = ((const float4*)src)[j];
        sum = fmaf(f.x, f.x, sum);
        sum = fmaf(f.y, f.y, sum);
        sum = fmaf(f.z, f.z, sum);
        sum = fmaf(f.w, f.w, sum);
        pk.s[j * 4 + 0] = f2bf_rne(f.x);
        pk.s[j * 4 + 1] = f2bf_rne(f.y);
        pk.s[j * 4 + 2] = f2bf_rne(f.z);
        pk.s[j * 4 + 3] = f2bf_rne(f.w);
    }
    ((uint4*)dst)[0] = pk.q[0];
    ((uint4*)dst)[1] = pk.q[1];

#pragma unroll
    for (int off = 16; off >= 1; off >>= 1)
        sum += __shfl_xor(sum, off, 32);
    if (lane == 0) cbsq[row] = sum;
}

// ---------------------------------------------------------------------------
// Phase 2: WMMA GEMM + running argmin + gather.
// ---------------------------------------------------------------------------
__global__ __launch_bounds__(256)
void vq_main(const float* __restrict__ X,
             const float* __restrict__ CB,
             const unsigned short* __restrict__ CBbf,
             const float* __restrict__ cbsq,
             float* __restrict__ outZq,
             float* __restrict__ outZt,
             float* __restrict__ outIdx)
{
    extern __shared__ __align__(128) unsigned char smemRaw[];
    unsigned short* ldsB0 = (unsigned short*)smemRaw;             // 2x128KB
    float*          ldsSq0 = (float*)(smemRaw + 2 * LDSB_ELEMS * 2);  // 2x512B
    LDS_AS int* ldsIx = (LDS_AS int*)(void*)
        (smemRaw + 2 * LDSB_ELEMS * 2 + 2 * NCHUNK * 4);

    const int tid  = threadIdx.x;
    const int wave = tid >> 5;
    const int lane = tid & 31;
    const int half = lane >> 4;   // 0: lanes 0-15, 1: lanes 16-31
    const int l16  = lane & 15;

    const int mBlock = blockIdx.x * MTILE;
    const int mWave  = mBlock + wave * 32;        // 32 rows per wave

    // ---- Two resident A tile-sets (rows mWave..+15 and mWave+16..+31).
    // 16-bit A layout: lane = M%16 (+16 dup), VGPR0-7: K=16*half..+15
    // (contiguous) within each 32-wide K slice, matching B.
    FragBF aLo[16], aHi[16];
#pragma unroll
    for (int setI = 0; setI < 2; ++setI) {
        FragBF* a = setI ? aHi : aLo;
        const float* xr = X + (size_t)(mWave + setI * 16 + l16) * CDIM;
#pragma unroll
        for (int cc = 0; cc < 16; ++cc) {
            const int c0 = cc * 32;
            const float4* p0 = (const float4*)(xr + c0 + 16 * half);
            const float4* p1 = (const float4*)(xr + c0 + 16 * half + 8);
            float4 f0 = p0[0], f1 = p0[1];
            float4 f2 = p1[0], f3 = p1[1];
            float t[16] = { f0.x, f0.y, f0.z, f0.w,  f1.x, f1.y, f1.z, f1.w,
                            f2.x, f2.y, f2.z, f2.w,  f3.x, f3.y, f3.z, f3.w };
#pragma unroll
            for (int i = 0; i < 16; ++i) a[cc].s[i] = f2bf_rne(t[i]);
        }
    }

    float bestV0[8], bestV1[8];
    int   bestI0[8], bestI1[8];
#pragma unroll
    for (int r = 0; r < 8; ++r) {
        bestV0[r] = 3.4028235e38f; bestI0[r] = 0;
        bestV1[r] = 3.4028235e38f; bestI1[r] = 0;
    }

    // ---- Prologue: start chunk 0 into buffer 0.
    issue_chunk_copy((const v4i*)CBbf, (v4i*)ldsB0, tid);
    if (tid < NCHUNK)
        ((LDS_AS float*)(void*)ldsSq0)[tid] = cbsq[tid];

#pragma unroll 1
    for (int c = 0; c < NCHUNKS; ++c) {
        const int buf = c & 1;
        const bool more = (c + 1) < NCHUNKS;
        if (more) {
            // Overwrites the buffer last READ in iteration c-1 (barriered).
            issue_chunk_copy((const v4i*)(CBbf + (size_t)(c + 1) * LDSB_ELEMS),
                             (v4i*)(ldsB0 + (size_t)(buf ^ 1) * LDSB_ELEMS), tid);
            if (tid < NCHUNK)
                ((LDS_AS float*)(void*)(ldsSq0 + (buf ^ 1) * NCHUNK))[tid] =
                    cbsq[(c + 1) * NCHUNK + tid];
        }
        wait_chunk_copy(more);
        __syncthreads();   // publish buffer `buf`

        // Explicit addrspace(3) -> guaranteed ds_load_b128 path.
        const LDS_AS unsigned short* bb =
            (const LDS_AS unsigned short*)(void*)(ldsB0 + (size_t)buf * LDSB_ELEMS);
        const LDS_AS float* sqb =
            (const LDS_AS float*)(void*)(ldsSq0 + buf * NCHUNK);

#pragma unroll 1
        for (int t = 0; t < NCHUNK / 16; ++t) {
            const int nLocal = t * 16 + l16;     // B column = codebook row
            const LDS_AS unsigned short* brow = bb + nLocal * CDIM + 16 * half;

            // 4-slot rotating B buffer: 3 DS double-loads always in flight
            // ahead of the consuming wmma pair (no full dscnt drains).
            FragBF b[4];
#pragma unroll
            for (int p = 0; p < 4; ++p)
                b[p].v = *(const LDS_AS v16bf*)(brow + p * 32);

            v8f acc0 = {};
            v8f acc1 = {};
#pragma unroll
            for (int cc = 0; cc < 16; ++cc) {
                const int slot = cc & 3;
                // B layout: lane = N%16 (+16), 16 contiguous K per lane;
                // one aligned 32B LDS read feeds TWO wmmas (M 0-15 / 16-31).
                acc0 = __builtin_amdgcn_wmma_f32_16x16x32_bf16(
                           false, aLo[cc].v, false, b[slot].v, (short)0, acc0, false, false);
                acc1 = __builtin_amdgcn_wmma_f32_16x16x32_bf16(
                           false, aHi[cc].v, false, b[slot].v, (short)0, acc1, false, false);
                if (cc + 4 < 16)
                    b[slot].v = *(const LDS_AS v16bf*)(brow + (cc + 4) * 32);
            }
            const float sq   = sqb[t * 16 + l16];
            const int   nIdx = c * NCHUNK + t * 16 + l16;
#pragma unroll
            for (int r = 0; r < 8; ++r) {
                const float s0 = fmaf(2.0f, acc0[r], sq);   // ||c||^2 + 2 x.c
                if (s0 < bestV0[r]) { bestV0[r] = s0; bestI0[r] = nIdx; }
                else if (s0 == bestV0[r] && nIdx < bestI0[r]) { bestI0[r] = nIdx; }
                const float s1 = fmaf(2.0f, acc1[r], sq);
                if (s1 < bestV1[r]) { bestV1[r] = s1; bestI1[r] = nIdx; }
                else if (s1 == bestV1[r] && nIdx < bestI1[r]) { bestI1[r] = nIdx; }
            }
        }
        __syncthreads();   // done reading buffer `buf` -> safe to refill later
    }

    // ---- Reduce argmin across the 16 lanes sharing each M row.
    // C/D layout: VGPR r, lanes 0-15 -> M=r, lanes 16-31 -> M=r+8; N=lane%16.
#pragma unroll
    for (int setI = 0; setI < 2; ++setI) {
        float* bv = setI ? bestV1 : bestV0;
        int*   bi = setI ? bestI1 : bestI0;
#pragma unroll
        for (int r = 0; r < 8; ++r) {
            float v = bv[r];
            int   i = bi[r];
#pragma unroll
            for (int off = 1; off < 16; off <<= 1) {
                const float ov = __shfl_xor(v, off, 32);
                const int   oi = __shfl_xor(i, off, 32);
                if (ov < v || (ov == v && oi < i)) { v = ov; i = oi; }
            }
            if (l16 == 0)
                ldsIx[wave * 32 + setI * 16 + half * 8 + r] = i;
        }
    }
    __syncthreads();

    // ---- Emit indices (as float) and gather winning codebook rows (fp32).
    const int cbRow = ldsIx[tid];                 // one row per thread
    outIdx[mBlock + tid] = (float)cbRow;

    const float4* src = (const float4*)(CB + (size_t)cbRow * CDIM);
    float4* d0 = (float4*)(outZq + (size_t)(mBlock + tid) * CDIM);
    float4* d1 = (float4*)(outZt + (size_t)(mBlock + tid) * CDIM);
#pragma unroll 4
    for (int j = 0; j < CDIM / 4; ++j) {
        const float4 v = src[j];
        d0[j] = v;
        d1[j] = v;
    }
}

// ---------------------------------------------------------------------------
extern "C" void kernel_launch(void* const* d_in, const int* in_sizes, int n_in,
                              void* d_out, int out_size, void* d_ws, size_t ws_size,
                              hipStream_t stream)
{
    const float* X  = (const float*)d_in[0];   // z_e_x  [8,4096,512]
    const float* CB = (const float*)d_in[1];   // codebook [8192,512]

    unsigned short* CBbf = (unsigned short*)d_ws;                           // 8 MB
    float*          cbsq = (float*)((char*)d_ws + (size_t)KDIM * CDIM * 2); // 32 KB

    float* outZq  = (float*)d_out;                   // 16,777,216 floats
    float* outZt  = outZq + (size_t)BN_TOTAL * CDIM; // 16,777,216 floats
    float* outIdx = outZt + (size_t)BN_TOTAL * CDIM; // 32,768 floats

    // Phase 1: one wave per codebook row -> 8192 waves = 1024 blocks of 256.
    vq_prep_cb<<<(KDIM * 32) / 256, 256, 0, stream>>>(CB, CBbf, cbsq);

    // Phase 2: 128 blocks (~1 per WGP) x 256 threads; ~258KB dynamic LDS
    // (double-buffered 2x128KB B staging, legal on a 320KB-LDS WGP).
    const size_t smem = 2 * (size_t)LDSB_ELEMS * 2 + 2 * NCHUNK * 4 + MTILE * 4;
    vq_main<<<BN_TOTAL / MTILE, 256, smem, stream>>>(X, CB, CBbf, cbsq,
                                                     outZq, outZt, outIdx);
}
